// FlashAttention_16286515986958
// MI455X (gfx1250) — matrix operand order repaired and checked by reading, not compiled, not run on hardware
//
#include <hip/hip_runtime.h>
#include <hip/hip_bf16.h>

typedef __bf16 bf16;
typedef __attribute__((ext_vector_type(16))) __bf16 v16bf;
typedef __attribute__((ext_vector_type(8)))  float  v8f;
typedef __attribute__((ext_vector_type(4)))  unsigned int u32x4;

// ---------------- helpers ----------------

__device__ __forceinline__ bf16 f2bf(float f) {
    unsigned int u = __builtin_bit_cast(unsigned int, f);
    unsigned int r = (u + 0x7FFFu + ((u >> 16) & 1u)) >> 16;   // RNE
    unsigned short s = (unsigned short)r;
    return __builtin_bit_cast(bf16, s);
}

__device__ __forceinline__ v8f wmma_bf16(v16bf a, v16bf b, v8f c) {
    // D = A(16x32) * B(32x16) + C, f32 accumulate
    return __builtin_amdgcn_wmma_f32_16x16x32_bf16(false, a, false, b,
                                                   (short)0, c, false, false);
}

// A fragment (16x32 bf16, row-major source, row stride ld elements).
// lane m = lane&15 holds row m; hi-half selects K sub-ranges per ISA layout:
//   hi=0: K 0..7 and 16..23 ; hi=1: K 8..15 and 24..31  (two 16B runs)
__device__ __forceinline__ v16bf load_a_frag(const bf16* base, int ld) {
    int lane = threadIdx.x & 31;
    int m = lane & 15, hi = lane >> 4;
    const bf16* row = base + (size_t)m * ld;
    union { u32x4 u[2]; v16bf f; } cv;
    cv.u[0] = *(const u32x4*)(row + 8 * hi);
    cv.u[1] = *(const u32x4*)(row + 16 + 8 * hi);
    return cv.f;
}

// B fragment (32x16 bf16) from a TRANSPOSED source: source row n holds the
// K-run for output column n (contiguous). lanes 0-15: K 0..7,16..23; lanes 16-31: K 8..15,24..31.
__device__ __forceinline__ v16bf load_bt_frag(const bf16* base, int ld) {
    int lane = threadIdx.x & 31;
    int n = lane & 15, hi = lane >> 4;
    const bf16* p = base + (size_t)n * ld + 8 * hi;
    union { u32x4 u[2]; v16bf f; } cv;
    cv.u[0] = *(const u32x4*)(p);
    cv.u[1] = *(const u32x4*)(p + 16);
    return cv.f;
}

__device__ __forceinline__ float half_max(float v) {
#pragma unroll
    for (int m = 1; m < 16; m <<= 1) v = fmaxf(v, __shfl_xor(v, m, 32));
    return v;
}
__device__ __forceinline__ float half_sum(float v) {
#pragma unroll
    for (int m = 1; m < 16; m <<= 1) v += __shfl_xor(v, m, 32);
    return v;
}

// ---------------- constants ----------------
// B=4, T=1024, C=1024, H=16, D=64
#define BT   4096      // B*T rows
#define CDIM 1024
#define C3   3072
#define TLEN 1024
#define HH   16
#define DD   64

// ---------------- kernel 0: fp32 -> bf16 elementwise ----------------
__global__ __launch_bounds__(256) void cvt_bf16_kernel(const float* __restrict__ in,
                                                       bf16* __restrict__ out, int n) {
    int i = blockIdx.x * 256 + threadIdx.x;
    if (i < n) out[i] = f2bf(in[i]);
}

// ---------------- kernel 1: transpose + convert  W[K][N] -> Wt[N][K] ----------------
__global__ __launch_bounds__(256) void cvt_transpose_kernel(const float* __restrict__ in,
                                                            bf16* __restrict__ out,
                                                            int K, int N) {
    int i = blockIdx.x * 256 + threadIdx.x;
    if (i < K * N) {
        int k = i / N, n = i - k * N;
        out[(size_t)n * K + k] = f2bf(in[i]);
    }
}

// ---------------- kernel 2: QKV GEMM + head scatter ----------------
// Register-blocked 32(M)x64(N) per wave: 2 A-frags x 4 B-frags -> 8 WMMA / k-step.
// Every 64-wide N strip lies in one q/k/v section and one head (d0 == 0).
// xbf[BT][C] @ wT[3C][C]^T + b -> Q[b,h,t,d]*0.125, K[b,h,t,d], Vt[b,h,d,t]  (bf16)
__global__ __launch_bounds__(256) void qkv_gemm_kernel(const bf16* __restrict__ xbf,
                                                       const bf16* __restrict__ wT,
                                                       const float* __restrict__ bias,
                                                       bf16* __restrict__ Q,
                                                       bf16* __restrict__ Kd,
                                                       bf16* __restrict__ Vt) {
    int wid  = threadIdx.x >> 5;
    int tile = blockIdx.x * 8 + wid;          // 128 Mstrips x 48 Nstrips = 6144 waves
    int mt = tile / 48, nt = tile - mt * 48;
    int m0 = mt * 32, c0 = nt * 64;

    int lane = threadIdx.x & 31;
    int n = lane & 15, hi = lane >> 4;

    v8f acc[2][4] = {};
    for (int k0 = 0; k0 < CDIM; k0 += 32) {
        v16bf a0 = load_a_frag(xbf + (size_t)m0 * CDIM + k0, CDIM);
        v16bf a1 = load_a_frag(xbf + (size_t)(m0 + 16) * CDIM + k0, CDIM);
#pragma unroll
        for (int ni = 0; ni < 4; ni++) {
            v16bf b = load_bt_frag(wT + (size_t)(c0 + 16 * ni) * CDIM + k0, CDIM);
            acc[0][ni] = wmma_bf16(a0, b, acc[0][ni]);
            acc[1][ni] = wmma_bf16(a1, b, acc[1][ni]);
        }
    }
#pragma unroll
    for (int ni = 0; ni < 4; ni++) {
        float bv = bias[c0 + 16 * ni + n];
#pragma unroll
        for (int r = 0; r < 8; r++) { acc[0][ni][r] += bv; acc[1][ni][r] += bv; }
    }

    int sec = c0 >> 10;          // 0=Q 1=K 2=V (strip never straddles sections)
    int cc  = c0 & 1023;
    int h   = cc >> 6;           // d0 == 0 always (64-wide strip == head dim)
    int bb  = m0 >> 10, t0 = m0 & 1023;
    size_t bh = (size_t)(bb * HH + h);

    if (sec == 0) {
#pragma unroll
        for (int mi = 0; mi < 2; mi++)
#pragma unroll
            for (int ni = 0; ni < 4; ni++)
#pragma unroll
                for (int r = 0; r < 8; r++) {
                    int t = t0 + 16 * mi + r + 8 * hi;
                    Q[(bh * TLEN + t) * DD + 16 * ni + n] =
                        f2bf(acc[mi][ni][r] * 0.125f);           // 1/sqrt(64)
                }
    } else if (sec == 1) {
#pragma unroll
        for (int mi = 0; mi < 2; mi++)
#pragma unroll
            for (int ni = 0; ni < 4; ni++)
#pragma unroll
                for (int r = 0; r < 8; r++) {
                    int t = t0 + 16 * mi + r + 8 * hi;
                    Kd[(bh * TLEN + t) * DD + 16 * ni + n] = f2bf(acc[mi][ni][r]);
                }
    } else {
#pragma unroll
        for (int mi = 0; mi < 2; mi++)
#pragma unroll
            for (int ni = 0; ni < 4; ni++)
#pragma unroll
                for (int r = 0; r < 8; r++) {
                    int t = t0 + 16 * mi + r + 8 * hi;
                    Vt[(bh * DD + 16 * ni + n) * TLEN + t] = f2bf(acc[mi][ni][r]);
                }
    }
}

// ---------------- kernel 3: causal flash attention ----------------
// per wave: one 16-query tile of one (b,h). Online softmax over 32-key blocks.
#define LPAD 40   // 80-byte row stride (16B aligned) for LDS P staging
__global__ __launch_bounds__(128) void attn_kernel(const bf16* __restrict__ Q,
                                                   const bf16* __restrict__ Kd,
                                                   const bf16* __restrict__ Vt,
                                                   bf16* __restrict__ Y) {
    __shared__ bf16 plds[4 * 16 * LPAD];
    int wid  = threadIdx.x >> 5;
    int lane = threadIdx.x & 31;
    int n = lane & 15, hi = lane >> 4;

    int gid = blockIdx.x * 4 + wid;           // 4096 q-tiles
    int b  = gid >> 10;
    int h  = (gid >> 6) & 15;
    int q0 = (gid & 63) * 16;

    size_t bh = (size_t)(b * HH + h);
    const bf16* Qb = Q  + bh * TLEN * DD;
    const bf16* Kb = Kd + bh * TLEN * DD;
    const bf16* Vb = Vt + bh * DD * TLEN;
    bf16* lw = plds + wid * 16 * LPAD;

    v16bf qa0 = load_a_frag(Qb + (size_t)q0 * DD, DD);       // d 0..31
    v16bf qa1 = load_a_frag(Qb + (size_t)q0 * DD + 32, DD);  // d 32..63

    float mi[8], li[8];
    v8f o0 = {}, o1 = {}, o2 = {}, o3 = {};
#pragma unroll
    for (int r = 0; r < 8; r++) { mi[r] = -1e30f; li[r] = 0.f; }

    int nkb = (q0 + 15) / 32 + 1;
    for (int kb = 0; kb < nkb; kb++) {
        int kbase = kb * 32;

        // S(16x32) = Qtile(16x64) @ K^T : two 16x16 n-tiles, K-dim split 32+32
        v8f s0 = {}, s1 = {};
        {
            v16bf k00 = load_bt_frag(Kb + (size_t)kbase * DD, DD);
            v16bf k10 = load_bt_frag(Kb + (size_t)(kbase + 16) * DD, DD);
            s0 = wmma_bf16(qa0, k00, s0);
            s1 = wmma_bf16(qa0, k10, s1);
            v16bf k01 = load_bt_frag(Kb + (size_t)kbase * DD + 32, DD);
            v16bf k11 = load_bt_frag(Kb + (size_t)(kbase + 16) * DD + 32, DD);
            s0 = wmma_bf16(qa1, k01, s0);
            s1 = wmma_bf16(qa1, k11, s1);
        }

        if (kbase + 31 > q0) {   // causal mask (partial block)
#pragma unroll
            for (int r = 0; r < 8; r++) {
                int qrow = q0 + r + 8 * hi;
                if (kbase + n      > qrow) s0[r] = -1e30f;
                if (kbase + 16 + n > qrow) s1[r] = -1e30f;
            }
        }

        // online softmax: rows live at (vgpr r, lane-half hi); reduce over 16 lanes
        float p0[8], p1[8];
#pragma unroll
        for (int r = 0; r < 8; r++) {
            float tm   = half_max(fmaxf(s0[r], s1[r]));
            float mnew = fmaxf(mi[r], tm);
            float alpha = __expf(mi[r] - mnew);
            p0[r] = __expf(s0[r] - mnew);
            p1[r] = __expf(s1[r] - mnew);
            float ls = half_sum(p0[r] + p1[r]);
            li[r] = li[r] * alpha + ls;
            mi[r] = mnew;
            o0[r] *= alpha; o1[r] *= alpha; o2[r] *= alpha; o3[r] *= alpha;
        }

        // C-layout -> A-layout via LDS (P is 16x32 bf16)
#pragma unroll
        for (int r = 0; r < 8; r++) {
            int row = r + 8 * hi;
            lw[row * LPAD + n]      = f2bf(p0[r]);
            lw[row * LPAD + 16 + n] = f2bf(p1[r]);
        }
        asm volatile("s_wait_dscnt 0x0" ::: "memory");
        v16bf pa;
        {
            const bf16* rowp = lw + (size_t)(lane & 15) * LPAD;
            union { u32x4 u[2]; v16bf f; } cv;
            cv.u[0] = *(const u32x4*)(rowp + 8 * hi);
            cv.u[1] = *(const u32x4*)(rowp + 16 + 8 * hi);
            pa = cv.f;
        }
        asm volatile("s_wait_dscnt 0x0" ::: "memory");

        // O(16x64) += P(16x32) @ V(32x64) : Vt rows are d, contiguous over keys
        v16bf v0 = load_bt_frag(Vb + kbase, TLEN);
        o0 = wmma_bf16(pa, v0, o0);
        v16bf v1 = load_bt_frag(Vb + (size_t)16 * TLEN + kbase, TLEN);
        o1 = wmma_bf16(pa, v1, o1);
        v16bf v2 = load_bt_frag(Vb + (size_t)32 * TLEN + kbase, TLEN);
        o2 = wmma_bf16(pa, v2, o2);
        v16bf v3 = load_bt_frag(Vb + (size_t)48 * TLEN + kbase, TLEN);
        o3 = wmma_bf16(pa, v3, o3);
    }

    // normalize + store y[b, t, h*64 + d] (bf16, feeds the projection GEMM)
#pragma unroll
    for (int r = 0; r < 8; r++) {
        float inv = 1.0f / li[r];
        int t = q0 + r + 8 * hi;
        size_t base = ((size_t)b * TLEN + t) * CDIM + h * DD + n;
        Y[base]      = f2bf(o0[r] * inv);
        Y[base + 16] = f2bf(o1[r] * inv);
        Y[base + 32] = f2bf(o2[r] * inv);
        Y[base + 48] = f2bf(o3[r] * inv);
    }
}

// ---------------- kernel 4: output projection ----------------
// Register-blocked 32(M)x64(N) per wave, same scheme as QKV GEMM.
// ybf[BT][C] @ wprojT[C][C]^T + b_proj -> out f32 [BT][C]
__global__ __launch_bounds__(256) void proj_gemm_kernel(const bf16* __restrict__ ybf,
                                                        const bf16* __restrict__ wT,
                                                        const float* __restrict__ bias,
                                                        float* __restrict__ out) {
    int wid  = threadIdx.x >> 5;
    int tile = blockIdx.x * 8 + wid;          // 128 Mstrips x 16 Nstrips = 2048 waves
    int mt = tile >> 4, nt = tile & 15;
    int m0 = mt * 32, n0 = nt * 64;

    int lane = threadIdx.x & 31;
    int n = lane & 15, hi = lane >> 4;

    v8f acc[2][4] = {};
    for (int k0 = 0; k0 < CDIM; k0 += 32) {
        v16bf a0 = load_a_frag(ybf + (size_t)m0 * CDIM + k0, CDIM);
        v16bf a1 = load_a_frag(ybf + (size_t)(m0 + 16) * CDIM + k0, CDIM);
#pragma unroll
        for (int ni = 0; ni < 4; ni++) {
            v16bf b = load_bt_frag(wT + (size_t)(n0 + 16 * ni) * CDIM + k0, CDIM);
            acc[0][ni] = wmma_bf16(a0, b, acc[0][ni]);
            acc[1][ni] = wmma_bf16(a1, b, acc[1][ni]);
        }
    }
#pragma unroll
    for (int ni = 0; ni < 4; ni++) {
        float bv = bias[n0 + 16 * ni + n];
#pragma unroll
        for (int mi = 0; mi < 2; mi++)
#pragma unroll
            for (int r = 0; r < 8; r++) {
                out[(size_t)(m0 + 16 * mi + r + 8 * hi) * CDIM + n0 + 16 * ni + n] =
                    acc[mi][ni][r] + bv;
            }
    }
}

// ---------------- launcher ----------------
extern "C" void kernel_launch(void* const* d_in, const int* in_sizes, int n_in,
                              void* d_out, int out_size, void* d_ws, size_t ws_size,
                              hipStream_t stream) {
    const float* x      = (const float*)d_in[0];
    const float* w_attn = (const float*)d_in[1];
    const float* b_attn = (const float*)d_in[2];
    const float* w_proj = (const float*)d_in[3];
    const float* b_proj = (const float*)d_in[4];
    float* out = (float*)d_out;

    char* ws = (char*)d_ws;
    bf16* xbf    = (bf16*)(ws);                                   // 8 MB
    bf16* wattnT = (bf16*)(ws + 8388608);                         // 6 MB
    bf16* wprojT = (bf16*)(ws + 8388608 + 6291456);               // 2 MB
    bf16* Qbf    = (bf16*)(ws + 16777216);                        // 8 MB
    bf16* Kbf    = (bf16*)(ws + 25165824);                        // 8 MB
    bf16* Vt     = (bf16*)(ws + 33554432);                        // 8 MB
    bf16* ybf    = (bf16*)(ws + 41943040);                        // 8 MB

    cvt_bf16_kernel<<<(BT * CDIM) / 256, 256, 0, stream>>>(x, xbf, BT * CDIM);
    cvt_transpose_kernel<<<(CDIM * C3) / 256, 256, 0, stream>>>(w_attn, wattnT, CDIM, C3);
    cvt_transpose_kernel<<<(CDIM * CDIM) / 256, 256, 0, stream>>>(w_proj, wprojT, CDIM, CDIM);

    qkv_gemm_kernel<<<(128 * 48) / 8, 256, 0, stream>>>(xbf, wattnT, b_attn, Qbf, Kbf, Vt);
    attn_kernel<<<4096 / 4, 128, 0, stream>>>(Qbf, Kbf, Vt, ybf);
    proj_gemm_kernel<<<(128 * 16) / 8, 256, 0, stream>>>(ybf, wprojT, b_proj, out);
}
